// RoMaTransformer_78829829750926
// MI455X (gfx1250) — matrix operand order repaired and checked by reading, not compiled, not run on hardware
//
#include <hip/hip_runtime.h>
#include <cstdint>
#include <cstddef>

// ---------------------------------------------------------------------------
// RoMa-style transformer for MI455X (gfx1250), wave32 + WMMA f16->f32.
// All GEMMs (QKV/O/FFN/proj/anchor) and both attention matmuls run on
// v_wmma_f32_16x16x32_f16.  Attention is fused flash-style so the 1024x1024
// score matrix (and the rank-1 vessel bias) is never materialized.
// Weights are pre-transposed to [N][K] f16 so all LDS staging is b128 copies.
// ---------------------------------------------------------------------------

typedef _Float16 half_t;
typedef __attribute__((ext_vector_type(16))) _Float16 v16h;
typedef __attribute__((ext_vector_type(8)))  _Float16 v8h;
typedef __attribute__((ext_vector_type(8)))  float    v8f;

static constexpr int kD    = 256;   // model dim
static constexpr int kN    = 1024;  // tokens per image (32x32)
static constexpr int kS    = 4;     // slices: img*2 + batch
static constexpr int kM    = 4096;  // kS * kN
static constexpr int kDFF  = 1024;
static constexpr int kANCH = 4096;
static constexpr int kH    = 8;     // heads
static constexpr int kDH   = 32;    // head dim == WMMA K

#define ASTR 40   // padded LDS stride (halves) for 32-wide K tiles
#define BSTR 40
#define KSTR 40
#define VSTR 72   // padded LDS stride for 64-wide key tiles
#define PSTR 72

// Build a v16h fragment from two 16-byte chunks (ds_load_b128 / global b128).
__device__ __forceinline__ v16h ld16(const half_t* p0, const half_t* p1) {
  v8h a = *(const v8h*)p0;
  v8h b = *(const v8h*)p1;
  return __builtin_shufflevector(a, b, 0,1,2,3,4,5,6,7,8,9,10,11,12,13,14,15);
}
__device__ __forceinline__ v8f wmma16(v16h a, v16h b, v8f c) {
  // (neg_a, A, neg_b, B, c_mod, C, reuse_a, reuse_b)
  return __builtin_amdgcn_wmma_f32_16x16x32_f16(false, a, false, b, (short)0, c,
                                                false, false);
}
__device__ __forceinline__ v8f vzero8() {
  v8f z = {0.f,0.f,0.f,0.f,0.f,0.f,0.f,0.f};
  return z;
}

// ------------------------------- elementwise -------------------------------
__global__ void k_cvt(const float* __restrict__ s, half_t* __restrict__ d, int n) {
  int i = blockIdx.x * 256 + threadIdx.x;
  if (i < n) d[i] = (half_t)s[i];
}
// f32 [K][N] row-major  ->  f16 [N][K] (transposed), one-time weight prep.
__global__ void k_cvtT(const float* __restrict__ s, half_t* __restrict__ d,
                       int K, int N) {
  int i = blockIdx.x * 256 + threadIdx.x;
  if (i < K * N) {
    int k = i / N, n = i - k * N;
    d[(size_t)n * K + k] = (half_t)s[i];
  }
}
__global__ void k_copy(const float* __restrict__ s, float* __restrict__ d, int n) {
  int i = blockIdx.x * 256 + threadIdx.x;
  if (i < n) d[i] = s[i];
}

// ------------------------ sinusoidal positional enc ------------------------
__global__ void k_pos(float* __restrict__ pos) {
  int t = blockIdx.x, d = threadIdx.x;
  int hh = t >> 5, ww = t & 31;
  const float two_pi = 6.283185307179586f;
  float coord; int dd;
  if (d < 128) { coord = (float)hh / (32.0f + 1e-6f) * two_pi; dd = d; }
  else         { coord = (float)ww / (32.0f + 1e-6f) * two_pi; dd = d - 128; }
  float ex  = (2.0f * (float)(dd >> 1)) / 128.0f;
  float ang = coord / powf(10000.0f, ex);
  pos[t * kD + d] = (dd & 1) ? cosf(ang) : sinf(ang);
}

// -------------------- bilinear mask downsample 256 -> 32 -------------------
__global__ void k_resize(const float* __restrict__ mask0, const float* __restrict__ mask1,
                         float* __restrict__ m0d, float* __restrict__ m1d) {
  int idx = blockIdx.x * 256 + threadIdx.x;          // 0..4095
  int which = idx >> 11;                             // 0: mask0, 1: mask1
  int r = idx & 2047;                                // b*1024 + t
  int b = r >> 10, t = r & 1023;
  int hh = t >> 5, ww = t & 31;
  const float* src = (which ? mask1 : mask0) + (size_t)b * 256 * 256;
  float y = (hh + 0.5f) * 8.0f - 0.5f;
  float x = (ww + 0.5f) * 8.0f - 0.5f;
  int y0 = (int)floorf(y), x0 = (int)floorf(x);
  float fy = y - y0, fx = x - x0;
  int y1 = y0 + 1 > 255 ? 255 : y0 + 1;
  int x1 = x0 + 1 > 255 ? 255 : x0 + 1;
  y0 = y0 < 0 ? 0 : y0;  x0 = x0 < 0 ? 0 : x0;
  float v = (1.f - fy) * ((1.f - fx) * src[y0 * 256 + x0] + fx * src[y0 * 256 + x1]) +
            fy         * ((1.f - fx) * src[y1 * 256 + x0] + fx * src[y1 * 256 + x1]);
  (which ? m1d : m0d)[r] = v;
}

// ------------------------------- layernorm ---------------------------------
__global__ __launch_bounds__(256) void k_ln(const float* __restrict__ X,
                                            const float* __restrict__ g,
                                            const float* __restrict__ b,
                                            half_t* __restrict__ Y) {
  __shared__ float red[256];
  int r = blockIdx.x, t = threadIdx.x;
  float x = X[(size_t)r * kD + t];
  red[t] = x; __syncthreads();
  for (int s = 128; s > 0; s >>= 1) { if (t < s) red[t] += red[t + s]; __syncthreads(); }
  float mean = red[0] * (1.0f / kD);
  __syncthreads();
  float d = x - mean;
  red[t] = d * d; __syncthreads();
  for (int s = 128; s > 0; s >>= 1) { if (t < s) red[t] += red[t + s]; __syncthreads(); }
  float var = red[0] * (1.0f / kD);
  Y[(size_t)r * kD + t] = (half_t)(d * rsqrtf(var + 1e-5f) * g[t] + b[t]);
}

// ------------------------------ generic GEMM -------------------------------
// C[M,N] = A[M,K](f16) @ W^T[N,K](f16, pre-transposed) + bias.
// Block tile 256x64, 8 waves; each wave: two 16-row frags x four 16-col tiles
// = 8 WMMAs per K-step.  Epilogue specialization via template flags.
template <bool RELU, bool RESID, bool OUTF, bool OUTH, bool OUTHT>
__global__ __launch_bounds__(256) void k_gemm(
    const half_t* __restrict__ A, const half_t* __restrict__ Wt,
    const float* __restrict__ bias, const float* __restrict__ resid,
    float* __restrict__ outF, half_t* __restrict__ outH,
    half_t* __restrict__ outHT, int M, int N, int K) {
  __shared__ __align__(16) half_t As[256 * ASTR];
  __shared__ __align__(16) half_t Bs[64 * BSTR];    // [n][k]
  int tid = threadIdx.x, lane = tid & 31, wave = tid >> 5;
  int hiq = lane >> 4, lo = lane & 15;
  int n0 = blockIdx.x * 64, m0 = blockIdx.y * 256;
  v8f acc[2][4];
#pragma unroll
  for (int mf = 0; mf < 2; mf++)
#pragma unroll
    for (int nt = 0; nt < 4; nt++) acc[mf][nt] = vzero8();

  for (int k0 = 0; k0 < K; k0 += 32) {
    __syncthreads();
    {  // stage A 256x32: one 64B row per thread, pure b128 copies
      const uint4* src = (const uint4*)(A + (size_t)(m0 + tid) * K + k0);
      uint4 d0 = src[0], d1 = src[1], d2 = src[2], d3 = src[3];
      uint4* dst = (uint4*)(As + tid * ASTR);
      dst[0] = d0; dst[1] = d1; dst[2] = d2; dst[3] = d3;
    }
    {  // stage Wt 64x32: pre-transposed, pure b128 copies
      int n = tid >> 2, ch = tid & 3;
      uint4 d0 = *(const uint4*)(Wt + (size_t)(n0 + n) * K + k0 + ch * 8);
      *(uint4*)(Bs + n * BSTR + ch * 8) = d0;
    }
    __syncthreads();
    const half_t* ar0 = As + (wave * 32 + lo) * ASTR;
    const half_t* ar1 = ar0 + 16 * ASTR;
    v16h a0 = ld16(ar0 + hiq * 8, ar0 + 16 + hiq * 8);
    v16h a1 = ld16(ar1 + hiq * 8, ar1 + 16 + hiq * 8);
#pragma unroll
    for (int nt = 0; nt < 4; nt++) {
      const half_t* bc = Bs + (nt * 16 + lo) * BSTR + hiq * 8;
      v16h b = ld16(bc, bc + 16);
      acc[0][nt] = wmma16(a0, b, acc[0][nt]);
      acc[1][nt] = wmma16(a1, b, acc[1][nt]);
    }
  }
#pragma unroll
  for (int mf = 0; mf < 2; mf++) {
#pragma unroll
    for (int nt = 0; nt < 4; nt++) {
      int col = n0 + nt * 16 + lo;
      float bv = bias[col];
#pragma unroll
      for (int v = 0; v < 8; v++) {
        int row = m0 + wave * 32 + mf * 16 + hiq * 8 + v;
        float val = acc[mf][nt][v] + bv;
        if (RELU) val = fmaxf(val, 0.0f);
        size_t o = (size_t)row * N + col;
        if (RESID) val += resid[o];
        if (OUTF) outF[o] = val;
        if (OUTH) outH[o] = (half_t)val;
        if (OUTHT) outHT[(size_t)col * M + row] = (half_t)val;  // [dim][token]
      }
    }
  }
}

// ----------------- 1x1 conv projection (A = feat^T) + pos ------------------
__global__ __launch_bounds__(256) void k_proj(
    const float* __restrict__ feat0, const float* __restrict__ feat1,
    const half_t* __restrict__ WtP, const float* __restrict__ bias,
    const float* __restrict__ pos, float* __restrict__ F) {
  __shared__ __align__(16) half_t As[128 * ASTR];
  __shared__ __align__(16) half_t Bs[64 * BSTR];
  int tid = threadIdx.x, lane = tid & 31, wave = tid >> 5;
  int hiq = lane >> 4, lo = lane & 15;
  int n0 = blockIdx.x * 64, t0 = blockIdx.y * 128, sl = blockIdx.z;
  const float* feat = ((sl >> 1) ? feat1 : feat0) + (size_t)(sl & 1) * kD * kN;
  v8f acc[4];
#pragma unroll
  for (int i = 0; i < 4; i++) acc[i] = vzero8();

  for (int k0 = 0; k0 < kD; k0 += 32) {
    __syncthreads();
    {  // A[t][c] = feat[c][t]: 32 channel rows of 128 contiguous tokens
      int kk = tid >> 3, c0 = (tid & 7) * 16;
      const float4* src = (const float4*)(feat + (size_t)(k0 + kk) * kN + t0 + c0);
#pragma unroll
      for (int q = 0; q < 4; q++) {
        float4 f = src[q];
        As[(c0 + q * 4 + 0) * ASTR + kk] = (half_t)f.x;
        As[(c0 + q * 4 + 1) * ASTR + kk] = (half_t)f.y;
        As[(c0 + q * 4 + 2) * ASTR + kk] = (half_t)f.z;
        As[(c0 + q * 4 + 3) * ASTR + kk] = (half_t)f.w;
      }
    }
    {  // pre-transposed proj weight
      int n = tid >> 2, ch = tid & 3;
      uint4 d0 = *(const uint4*)(WtP + (size_t)(n0 + n) * kD + k0 + ch * 8);
      *(uint4*)(Bs + n * BSTR + ch * 8) = d0;
    }
    __syncthreads();
    const half_t* ar = As + (wave * 16 + lo) * ASTR;
    v16h a = ld16(ar + hiq * 8, ar + 16 + hiq * 8);
#pragma unroll
    for (int nt = 0; nt < 4; nt++) {
      const half_t* bc = Bs + (nt * 16 + lo) * BSTR + hiq * 8;
      v16h b = ld16(bc, bc + 16);
      acc[nt] = wmma16(a, b, acc[nt]);
    }
  }
#pragma unroll
  for (int nt = 0; nt < 4; nt++) {
    int col = n0 + nt * 16 + lo;
    float bv = bias[col];
#pragma unroll
    for (int v = 0; v < 8; v++) {
      int tt = t0 + wave * 16 + hiq * 8 + v;
      F[(size_t)sl * kN * kD + (size_t)tt * kD + col] =
          acc[nt][v] + bv + pos[(size_t)tt * kD + col];
    }
  }
}

// --------------------------- fused flash attention -------------------------
// grid: (16 q-blocks, 8 heads, 4 slices), 128 threads (4 waves, 16 q-rows each).
// cross!=0: keys from opposite image, additive rank-1 bias qm[q]*km[k].
// V comes in pre-transposed ([dim][token]) so all LDS staging is b128 copies.
__global__ __launch_bounds__(128) void k_attn(
    const half_t* __restrict__ QH, const half_t* __restrict__ KH,
    const half_t* __restrict__ VHT, half_t* __restrict__ AOH,
    const float* __restrict__ m0d, const float* __restrict__ m1d, int cross) {
  __shared__ __align__(16) half_t Kl[64 * KSTR];      // [key][dim]
  __shared__ __align__(16) half_t Vt[32 * VSTR];      // [dim][key]
  __shared__ __align__(16) half_t Pl[4 * 16 * PSTR];  // per-wave P tiles [row][key]
  int tid = threadIdx.x, lane = tid & 31, wave = tid >> 5;
  int hiq = lane >> 4, lo = lane & 15;
  int qb = blockIdx.x, h = blockIdx.y, sl = blockIdx.z;
  int img = sl >> 1, b = sl & 1;
  int ks = cross ? ((1 - img) * 2 + b) : sl;
  const float* qm = nullptr; const float* km = nullptr;
  if (cross) {
    qm = (img == 0 ? m0d : m1d) + b * kN;
    km = (img == 0 ? m1d : m0d) + b * kN;
  }
  int q0 = qb * 64 + wave * 16;

  // Q fragment straight from global (A layout; rows = q0+lo, DH=32 = WMMA K)
  const half_t* qptr = QH + ((size_t)sl * kN + q0 + lo) * kD + h * kDH;
  v16h qa = ld16(qptr + hiq * 8, qptr + 16 + hiq * 8);

  float qmv[8];
#pragma unroll
  for (int v = 0; v < 8; v++) qmv[v] = cross ? qm[q0 + hiq * 8 + v] : 0.0f;

  float mrow[8], lrow[8];
#pragma unroll
  for (int v = 0; v < 8; v++) { mrow[v] = -1e30f; lrow[v] = 0.0f; }
  v8f o0 = vzero8(), o1 = vzero8();
  const float rsc = 0.17677669529663688f;  // 1/sqrt(32)
  half_t* Pw = Pl + wave * 16 * PSTR;

  for (int kt = 0; kt < kN; kt += 64) {
    __syncthreads();  // previous K/V tiles fully consumed
    {                 // stage K tile [64 keys][32 dims], b128 copies
      int key = tid >> 1, ch = tid & 1;
      const uint4* ksrc =
          (const uint4*)(KH + ((size_t)ks * kN + kt + key) * kD + h * kDH + ch * 16);
      uint4 k0v = ksrc[0], k1v = ksrc[1];
      uint4* kdst = (uint4*)(Kl + key * KSTR + ch * 16);
      kdst[0] = k0v; kdst[1] = k1v;
    }
    {  // stage V tile [32 dims][64 keys] from pre-transposed VHT, b128 copies
      int d = tid >> 2, ch = tid & 3;
      const uint4* vsrc =
          (const uint4*)(VHT + (size_t)(h * kDH + d) * kM + (size_t)ks * kN + kt + ch * 16);
      uint4 v0 = vsrc[0], v1 = vsrc[1];
      uint4* vdst = (uint4*)(Vt + d * VSTR + ch * 16);
      vdst[0] = v0; vdst[1] = v1;
    }
    __syncthreads();

    // scores: four 16x16 tiles against this 64-key chunk (one WMMA each)
    v8f sacc[4];
#pragma unroll
    for (int c = 0; c < 4; c++) {
      const half_t* kc = Kl + (c * 16 + lo) * KSTR + hiq * 8;
      v16h kb = ld16(kc, kc + 16);
      sacc[c] = wmma16(qa, kb, vzero8());
    }
    float s4[4][8];
#pragma unroll
    for (int c = 0; c < 4; c++) {
      float kmv = cross ? km[kt + c * 16 + lo] : 0.0f;
#pragma unroll
      for (int v = 0; v < 8; v++) {
        float x = sacc[c][v] * rsc;
        if (cross) x += qmv[v] * kmv;   // LAM == 1.0
        s4[c][v] = x;
      }
    }
    // online softmax: row reductions across the 16 lanes of each half
    float mn[8];
#pragma unroll
    for (int v = 0; v < 8; v++) {
      mn[v] = mrow[v];
#pragma unroll
      for (int c = 0; c < 4; c++) mn[v] = fmaxf(mn[v], s4[c][v]);
    }
#pragma unroll
    for (int msk = 1; msk < 16; msk <<= 1)
#pragma unroll
      for (int v = 0; v < 8; v++) mn[v] = fmaxf(mn[v], __shfl_xor(mn[v], msk, 32));
    float corr[8], rs[8];
#pragma unroll
    for (int v = 0; v < 8; v++) { corr[v] = __expf(mrow[v] - mn[v]); rs[v] = 0.0f; }
#pragma unroll
    for (int c = 0; c < 4; c++)
#pragma unroll
      for (int v = 0; v < 8; v++) {
        float p = __expf(s4[c][v] - mn[v]);
        rs[v] += p;
        Pw[(hiq * 8 + v) * PSTR + c * 16 + lo] = (half_t)p;  // C-layout -> LDS
      }
#pragma unroll
    for (int msk = 1; msk < 16; msk <<= 1)
#pragma unroll
      for (int v = 0; v < 8; v++) rs[v] += __shfl_xor(rs[v], msk, 32);
#pragma unroll
    for (int v = 0; v < 8; v++) {
      lrow[v] = lrow[v] * corr[v] + rs[v];
      mrow[v] = mn[v];
      o0[v] *= corr[v];
      o1[v] *= corr[v];
    }
    // P @ V : re-load P in A-layout (same-wave DS ops are in-order)
    const half_t* pr = Pw + lo * PSTR;
#pragma unroll
    for (int c2 = 0; c2 < 2; c2++) {
      v16h pa = ld16(pr + c2 * 32 + hiq * 8, pr + c2 * 32 + 16 + hiq * 8);
      const half_t* vb0 = Vt + lo * VSTR + c2 * 32 + hiq * 8;
      o0 = wmma16(pa, ld16(vb0, vb0 + 16), o0);
      const half_t* vb1 = Vt + (16 + lo) * VSTR + c2 * 32 + hiq * 8;
      o1 = wmma16(pa, ld16(vb1, vb1 + 16), o1);
    }
  }
#pragma unroll
  for (int v = 0; v < 8; v++) {
    int q = q0 + hiq * 8 + v;
    float inv = 1.0f / lrow[v];
    size_t base = ((size_t)sl * kN + q) * kD + h * kDH;
    AOH[base + lo] = (half_t)(o0[v] * inv);
    AOH[base + 16 + lo] = (half_t)(o1[v] * inv);
  }
}

// ------------------------- row softmax over 4096 ---------------------------
__global__ __launch_bounds__(256) void k_softmax(float* __restrict__ x) {
  __shared__ float red[256];
  int r = blockIdx.x, t = threadIdx.x;
  float* row = x + (size_t)r * kANCH;
  float vals[16];
  float mx = -1e30f;
#pragma unroll
  for (int i = 0; i < 16; i++) { vals[i] = row[t + i * 256]; mx = fmaxf(mx, vals[i]); }
  red[t] = mx; __syncthreads();
  for (int s = 128; s > 0; s >>= 1) { if (t < s) red[t] = fmaxf(red[t], red[t + s]); __syncthreads(); }
  mx = red[0]; __syncthreads();
  float sm = 0.f;
#pragma unroll
  for (int i = 0; i < 16; i++) { vals[i] = __expf(vals[i] - mx); sm += vals[i]; }
  red[t] = sm; __syncthreads();
  for (int s = 128; s > 0; s >>= 1) { if (t < s) red[t] += red[t + s]; __syncthreads(); }
  float inv = 1.0f / red[0];
#pragma unroll
  for (int i = 0; i < 16; i++) row[t + i * 256] = vals[i] * inv;
}

// ---------------------------------------------------------------------------
extern "C" void kernel_launch(void* const* d_in, const int* in_sizes, int n_in,
                              void* d_out, int out_size, void* d_ws, size_t ws_size,
                              hipStream_t stream) {
  (void)in_sizes; (void)n_in; (void)out_size; (void)ws_size;
  const float* feat0 = (const float*)d_in[0];
  const float* feat1 = (const float*)d_in[1];
  const float* mask0 = (const float*)d_in[2];
  const float* mask1 = (const float*)d_in[3];
  // params flattened jax-pytree style (sorted dict keys):
  // 4:anc1.b 5:anc1.w 6:anc2.b 7:anc2.w ; per layer l (base 8+26l):
  // +0 ck.b +1 ck.w +2 co.b +3 co.w +4 cq.b +5 cq.w +6 cv.b +7 cv.w
  // +8 f1.b +9 f1.w +10 f2.b +11 f2.w +12 ln1.b +13 ln1.g +14 ln2.b +15 ln2.g
  // +16 ln3.b +17 ln3.g +18 sk.b +19 sk.w +20 so.b +21 so.w +22 sq.b +23 sq.w
  // +24 sv.b +25 sv.w ; 112:proj.b 113:proj.w
  auto in = [&](int i) -> const float* { return (const float*)d_in[i]; };

  // ---- workspace carve (256B aligned) ----
  char* wsp = (char*)d_ws;
  auto carve = [&](size_t bytes) -> char* {
    char* p = wsp; wsp += (bytes + 255) & ~(size_t)255; return p;
  };
  float*  F   = (float*) carve((size_t)kM * kD * 4);   // residual stream [4,1024,256]
  half_t* XH  = (half_t*)carve((size_t)kM * kD * 2);   // normalized f16
  half_t* QH  = (half_t*)carve((size_t)kM * kD * 2);
  half_t* KH  = (half_t*)carve((size_t)kM * kD * 2);
  half_t* VHT = (half_t*)carve((size_t)kM * kD * 2);   // V transposed [256][4096]
  half_t* AOH = (half_t*)carve((size_t)kM * kD * 2);
  half_t* HH  = (half_t*)carve((size_t)kM * kDFF * 2);
  float*  POS = (float*) carve((size_t)kN * kD * 4);
  float*  M0d = (float*) carve(2048 * 4);
  float*  M1d = (float*) carve(2048 * 4);
  half_t* WH  = (half_t*)carve((size_t)5373952 * 2);   // all f16 weights (transposed)

  // ---- convert weights to f16, pre-transposed to [N][K] ----
  half_t* wp = WH;
  auto cvtw = [&](int idx, int K, int N) -> half_t* {
    half_t* p = wp; wp += (size_t)K * N;
    k_cvtT<<<dim3((K * N + 255) / 256), 256, 0, stream>>>(in(idx), p, K, N);
    return p;
  };
  half_t* Wproj = cvtw(113, kD, kD);
  half_t* Wanc1 = cvtw(5, kD, kD);
  half_t* Wanc2 = cvtw(7, kD, kANCH);
  half_t *Wsq[4], *Wsk[4], *Wsv[4], *Wso[4], *Wcq[4], *Wck[4], *Wcv[4], *Wco[4], *Wf1[4], *Wf2[4];
  for (int l = 0; l < 4; l++) {
    int lb = 8 + l * 26;
    Wck[l] = cvtw(lb + 1, kD, kD);    Wco[l] = cvtw(lb + 3, kD, kD);
    Wcq[l] = cvtw(lb + 5, kD, kD);    Wcv[l] = cvtw(lb + 7, kD, kD);
    Wf1[l] = cvtw(lb + 9, kD, kDFF);  Wf2[l] = cvtw(lb + 11, kDFF, kD);
    Wsk[l] = cvtw(lb + 19, kD, kD);   Wso[l] = cvtw(lb + 21, kD, kD);
    Wsq[l] = cvtw(lb + 23, kD, kD);   Wsv[l] = cvtw(lb + 25, kD, kD);
  }

  // ---- front end ----
  k_pos<<<dim3(kN), 256, 0, stream>>>(POS);
  k_resize<<<dim3(16), 256, 0, stream>>>(mask0, mask1, M0d, M1d);
  k_proj<<<dim3(kD / 64, kN / 128, kS), 256, 0, stream>>>(feat0, feat1, Wproj, in(112), POS, F);

  // epilogue-specialized GEMM wrappers
  auto gemm_h = [&](const half_t* A, const half_t* W, const float* bias,
                    half_t* oH, int M, int N, int K) {  // f16 out
    k_gemm<false, false, false, true, false>
        <<<dim3(N / 64, M / 256), 256, 0, stream>>>(A, W, bias, nullptr, nullptr, oH, nullptr, M, N, K);
  };
  auto gemm_ht = [&](const half_t* A, const half_t* W, const float* bias,
                     half_t* oHT, int M, int N, int K) {  // transposed f16 out
    k_gemm<false, false, false, false, true>
        <<<dim3(N / 64, M / 256), 256, 0, stream>>>(A, W, bias, nullptr, nullptr, nullptr, oHT, M, N, K);
  };
  auto gemm_relu_h = [&](const half_t* A, const half_t* W, const float* bias,
                         half_t* oH, int M, int N, int K) {  // relu + f16 out
    k_gemm<true, false, false, true, false>
        <<<dim3(N / 64, M / 256), 256, 0, stream>>>(A, W, bias, nullptr, nullptr, oH, nullptr, M, N, K);
  };
  auto gemm_resid = [&](const half_t* A, const half_t* W, const float* bias,
                        float* io, int M, int N, int K) {  // += residual, f32 out
    k_gemm<false, true, true, false, false>
        <<<dim3(N / 64, M / 256), 256, 0, stream>>>(A, W, bias, io, io, nullptr, nullptr, M, N, K);
  };
  auto gemm_f = [&](const half_t* A, const half_t* W, const float* bias,
                    float* oF, int M, int N, int K) {  // f32 out
    k_gemm<false, false, true, false, false>
        <<<dim3(N / 64, M / 256), 256, 0, stream>>>(A, W, bias, nullptr, oF, nullptr, nullptr, M, N, K);
  };

  // ---- transformer layers ----
  for (int l = 0; l < 4; l++) {
    int lb = 8 + l * 26;
    // self-attention (shared weights across all 4 slices)
    k_ln<<<dim3(kM), 256, 0, stream>>>(F, in(lb + 13), in(lb + 12), XH);
    gemm_h (XH, Wsq[l], in(lb + 22), QH, kM, kD, kD);
    gemm_h (XH, Wsk[l], in(lb + 18), KH, kM, kD, kD);
    gemm_ht(XH, Wsv[l], in(lb + 24), VHT, kM, kD, kD);
    k_attn<<<dim3(16, kH, kS), 128, 0, stream>>>(QH, KH, VHT, AOH, M0d, M1d, 0);
    gemm_resid(AOH, Wso[l], in(lb + 20), F, kM, kD, kD);
    // cross-attention with rank-1 vessel bias
    k_ln<<<dim3(kM), 256, 0, stream>>>(F, in(lb + 15), in(lb + 14), XH);
    gemm_h (XH, Wcq[l], in(lb + 4), QH, kM, kD, kD);
    gemm_h (XH, Wck[l], in(lb + 0), KH, kM, kD, kD);
    gemm_ht(XH, Wcv[l], in(lb + 6), VHT, kM, kD, kD);
    k_attn<<<dim3(16, kH, kS), 128, 0, stream>>>(QH, KH, VHT, AOH, M0d, M1d, 1);
    gemm_resid(AOH, Wco[l], in(lb + 2), F, kM, kD, kD);
    // FFN
    k_ln<<<dim3(kM), 256, 0, stream>>>(F, in(lb + 17), in(lb + 16), XH);
    gemm_relu_h(XH, Wf1[l], in(lb + 8), HH, kM, kDFF, kD);
    gemm_resid (HH, Wf2[l], in(lb + 10), F, kM, kD, kDFF);
  }

  // ---- anchor head on f0 (slices 0..1) ----
  k_cvt<<<dim3((2048 * kD + 255) / 256), 256, 0, stream>>>(F, XH, 2048 * kD);
  gemm_relu_h(XH, Wanc1, in(4), QH, 2048, kD, kD);
  float* logits = (float*)d_out;  // anchor_probs region [2,1024,4096]
  gemm_f(QH, Wanc2, in(6), logits, 2048, kANCH, kD);
  k_softmax<<<dim3(2048), 256, 0, stream>>>(logits);

  // ---- f0, f1 tail of d_out ----
  k_copy<<<dim3((kM * kD + 255) / 256), 256, 0, stream>>>(
      F, (float*)d_out + (size_t)2048 * kANCH, kM * kD);
}
